// DependencyParseModel_3805341024550
// MI455X (gfx1250) — matrix operand order repaired and checked by reading, not compiled, not run on hardware
//
#include <hip/hip_runtime.h>
#include <hip/hip_bf16.h>
#include <math.h>

typedef __attribute__((ext_vector_type(16))) __bf16 v16bf;
typedef __attribute__((ext_vector_type(8)))  float  v8f;

#define T_   256
#define H_   400
#define G4_  1600   // 4*H gates
#define BI_  800    // 2*H
#define MI_  1600   // 2*BI

__device__ __forceinline__ float fast_tanh(float x) {
#if __has_builtin(__builtin_amdgcn_tanhf)
    return __builtin_amdgcn_tanhf(x);     // gfx1250 v_tanh_f32
#else
    return tanhf(x);
#endif
}
__device__ __forceinline__ float fast_sig(float x) {
    return 1.f / (1.f + __expf(-x));
}

// ---------------- fp32 -> bf16 one-time weight conversion ----------------
__global__ void k_f2b(const float* __restrict__ in, __bf16* __restrict__ out, int n) {
    int i = blockIdx.x * 256 + threadIdx.x;
    if (i < n) out[i] = (__bf16)in[i];
}

// ---------------- transpose: (B,R,C) -> (B,C,R), fp32 (recurrent weights) ----
__global__ void k_transpose(const float* __restrict__ in, float* __restrict__ out,
                            int R, int C) {
    long b = blockIdx.y;
    int idx = blockIdx.x * 256 + threadIdx.x;
    if (idx >= R * C) return;
    int r = idx / C, c = idx % C;
    out[b * (long)R * C + (long)c * R + r] = in[b * (long)R * C + idx];
}

// ---------------- embedding gather -> bf16 x ----------------
__global__ void k_embed(const int* __restrict__ words, const int* __restrict__ tags,
                        const float* __restrict__ wemb, const float* __restrict__ temb,
                        __bf16* __restrict__ x) {
    int t = blockIdx.x, tid = threadIdx.x;
    if (tid < 400) {
        float v = (tid < 300) ? wemb[(long)words[t] * 300 + tid]
                              : temb[(long)tags[t] * 100 + (tid - 300)];
        x[(long)t * 400 + tid] = (__bf16)v;
    }
}

// ---------------- WMMA bf16 GEMM ----------------
// C[M=256,N] = A[256,K](bf16) @ W[N,K](bf16)^T (+b1+b2, opt tanh, fp32 or bf16 out)
// One wave computes a 16x64 strip (4 N-tiles) sharing one A fragment -> 4 WMMAs/A-load.
// Branchless inner loop: OOB W rows clamped (their columns are never stored).
union FragBF { v16bf v; uint4 q[2]; };

__global__ void k_gemm(const __bf16* __restrict__ A, int lda,
                       const __bf16* __restrict__ W, int ldw,
                       const float* __restrict__ b1, const float* __restrict__ b2,
                       float* __restrict__ Cf, __bf16* __restrict__ Cb, int ldc,
                       int N, int K, int do_tanh)
{
    const int lane = threadIdx.x & 31;
    const int wv   = threadIdx.x >> 5;
    const int g    = lane >> 4;          // K-half selector
    const int mr   = lane & 15;          // row (A) / output feature (B) within tile
    const int chunk = blockIdx.y * 8 + wv;
    if (chunk * 64 >= N) return;         // wave-uniform exit
    const int row = blockIdx.x * 16 + mr;
    const __bf16* arow = A + (long)row * lda;
    const __bf16* wrow[4];
    int ncol[4];
#pragma unroll
    for (int i = 0; i < 4; ++i) {
        int n = chunk * 64 + i * 16 + mr;
        ncol[i] = n;
        int nl = (n < N) ? n : (N - 1);  // clamp; OOB column results discarded at store
        wrow[i] = W + (long)nl * ldw;
    }
    v8f acc[4];
#pragma unroll
    for (int i = 0; i < 4; ++i)
#pragma unroll
        for (int r = 0; r < 8; ++r) acc[i][r] = 0.f;

    for (int k0 = 0; k0 < K; k0 += 32) {
        // A fragment (16x32): e=0..7 -> K=k0+g*8+e ; e=8..15 -> K=k0+16+g*8+(e-8)
        FragBF fa;
        fa.q[0] = *(const uint4*)(arow + k0 + g * 8);
        fa.q[1] = *(const uint4*)(arow + k0 + 16 + g * 8);
#pragma unroll
        for (int i = 0; i < 4; ++i) {
            // B fragment (32x16): e=0..7 -> K=k0+g*8+e ; e=8..15 -> K=k0+16+g*8+(e-8)
            FragBF fb;
            fb.q[0] = *(const uint4*)(wrow[i] + k0 + g * 8);
            fb.q[1] = *(const uint4*)(wrow[i] + k0 + 16 + g * 8);
            acc[i] = __builtin_amdgcn_wmma_f32_16x16x32_bf16(false, fa.v, false, fb.v,
                                                             (short)0, acc[i], false, false);
        }
    }
#pragma unroll
    for (int i = 0; i < 4; ++i) {
        if (ncol[i] < N) {
            float badd = (b1 ? b1[ncol[i]] : 0.f) + (b2 ? b2[ncol[i]] : 0.f);
#pragma unroll
            for (int r = 0; r < 8; ++r) {   // D: VGPR r -> row r+8g, col lane%16
                int ro = blockIdx.x * 16 + r + 8 * g;
                float v = acc[i][r] + badd;
                if (do_tanh) v = fast_tanh(v);
                if (Cb) Cb[(long)ro * ldc + ncol[i]] = (__bf16)v;
                else    Cf[(long)ro * ldc + ncol[i]] = v;
            }
        }
    }
}

// ---------------- sequential LSTM recurrence (one block per direction) ---------
__global__ void __launch_bounds__(1024)
k_lstm(const float* __restrict__ xg,    // (2, T, 1600) — includes bih+bhh
       const float* __restrict__ WT,    // (2, 400, 1600) — Whh transposed (k-major)
       const float* __restrict__ h0,    // (4,1,400)
       const float* __restrict__ c0,
       int layer,
       __bf16* __restrict__ hseqb)      // (T, 800) bf16: [fwd(400) | bwd(400)]
{
    __shared__ float sh[H_], sc[H_], sg[G4_];
    const int dir = blockIdx.x;
    const int tid = threadIdx.x;
    const int off = (layer * 2 + dir) * H_;
    if (tid < H_) { sh[tid] = h0[off + tid]; sc[tid] = c0[off + tid]; }
    __syncthreads();
    const float* wt  = WT + (long)dir * H_ * G4_;
    const float* xgd = xg + (long)dir * T_ * G4_;
    const int j1 = tid, j2 = tid + 1024;
    for (int t = 0; t < T_; ++t) {
        const int tt = dir ? (T_ - 1 - t) : t;
        float s1 = xgd[(long)tt * G4_ + j1];
        float s2 = (j2 < G4_) ? xgd[(long)tt * G4_ + j2] : 0.f;
        for (int k = 0; k < H_; ++k) {
            float hk = sh[k];                       // LDS broadcast
            const float* wr = wt + (long)k * G4_;   // coalesced across threads
            s1 += wr[j1] * hk;
            if (j2 < G4_) s2 += wr[j2] * hk;
        }
        sg[j1] = s1;
        if (j2 < G4_) sg[j2] = s2;
        __syncthreads();
        if (tid < H_) {
            float i  = fast_sig(sg[tid]);
            float f  = fast_sig(sg[H_ + tid]);
            float gg = fast_tanh(sg[2 * H_ + tid]);
            float o  = fast_sig(sg[3 * H_ + tid]);
            float c  = f * sc[tid] + i * gg;
            float h  = o * fast_tanh(c);
            sc[tid] = c; sh[tid] = h;
            hseqb[(long)tt * BI_ + dir * H_ + tid] = (__bf16)h;
        }
        __syncthreads();
    }
}

// ---------------- arc scorer epilogue: raw[i,j] = Wa2·tanh(A[i]+B[j]) + ba2 ----
__global__ void k_arc(const float* __restrict__ Ap,   // (256,1600), ba1 folded in
                      const float* __restrict__ Bp,   // (256,1600)
                      const float* __restrict__ Wa2, const float* __restrict__ ba2,
                      float* __restrict__ scores)     // (257,257)
{
    __shared__ float sA[MI_], sW[MI_];
    const int i = blockIdx.x, tid = threadIdx.x;
    for (int m = tid; m < MI_; m += 256) { sA[m] = Ap[(long)i * MI_ + m]; sW[m] = Wa2[m]; }
    __syncthreads();
    const int wv = tid >> 5, lane = tid & 31;
    const float b2 = ba2[0];
    for (int jb = 0; jb < T_; jb += 8) {
        const int j = jb + wv;
        const float* br = Bp + (long)j * MI_;
        float part = 0.f;
        for (int m = lane; m < MI_; m += 32)
            part += sW[m] * fast_tanh(sA[m] + br[m]);
        for (int o = 16; o > 0; o >>= 1) part += __shfl_xor(part, o, 32);
        if (lane == 0) {
            float raw = part + b2;
            if (i == j) raw = 0.f;                        // zero diagonal
            scores[(long)(i + 1) * 257 + (j + 1)] = raw;
        }
    }
}

__global__ void k_border(float* __restrict__ scores) {
    int t = blockIdx.x * 256 + threadIdx.x;
    if (t <= 256) {
        scores[t]             = (t == 0) ? 1.f : 0.f;   // row 0
        scores[(long)t * 257] = (t == 0) ? 1.f : 0.f;   // col 0
    }
}

// ---------------- label features (bf16): lf[t] = [hV[t] | hV[clip(head-1)]] ----
__global__ void k_lf(const __bf16* __restrict__ hV, const int* __restrict__ arcs,
                     __bf16* __restrict__ lf) {
    int t = blockIdx.x, tid = threadIdx.x;
    int idx = arcs[1 + t] - 1;
    idx = idx < 0 ? 0 : (idx > T_ - 1 ? T_ - 1 : idx);
    for (int u = tid; u < BI_; u += 256) {
        lf[(long)t * MI_ + u]       = hV[(long)t * BI_ + u];
        lf[(long)t * MI_ + BI_ + u] = hV[(long)idx * BI_ + u];
    }
}

__global__ void k_labfin(const float* __restrict__ lab, const int* __restrict__ arcs,
                         float* __restrict__ out) {
    int gid = blockIdx.x * 256 + threadIdx.x;
    if (gid >= T_ * 40) return;
    int t = gid / 40;
    out[gid] = (arcs[1 + t] == 0) ? 0.f : lab[gid];
}

extern "C" void kernel_launch(void* const* d_in, const int* in_sizes, int n_in,
                              void* d_out, int out_size, void* d_ws, size_t ws_size,
                              hipStream_t stream)
{
    (void)in_sizes; (void)n_in; (void)out_size; (void)ws_size;
    const int*   words = (const int*)d_in[0];
    const int*   tags  = (const int*)d_in[1];
    const int*   arcs  = (const int*)d_in[2];
    const float* h0    = (const float*)d_in[3];
    const float* c0    = (const float*)d_in[4];
    const float* wemb  = (const float*)d_in[5];
    const float* temb  = (const float*)d_in[6];
    const float* Wih0  = (const float*)d_in[7];
    const float* Whh0  = (const float*)d_in[8];
    const float* bih0  = (const float*)d_in[9];
    const float* bhh0  = (const float*)d_in[10];
    const float* Wih1  = (const float*)d_in[11];
    const float* Whh1  = (const float*)d_in[12];
    const float* bih1  = (const float*)d_in[13];
    const float* bhh1  = (const float*)d_in[14];
    const float* Wa1   = (const float*)d_in[15];
    const float* ba1   = (const float*)d_in[16];
    const float* Wa2   = (const float*)d_in[17];
    const float* ba2   = (const float*)d_in[18];
    const float* Wl1   = (const float*)d_in[19];
    const float* bl1   = (const float*)d_in[20];
    const float* Wl2   = (const float*)d_in[21];
    const float* bl2   = (const float*)d_in[22];
    float* outF = (float*)d_out;

    // ---- workspace carve-up: fp32 region, then bf16 region (32B-aligned) ----
    float* wsf = (float*)d_ws;
    float* xg0 = wsf; wsf += 2 * 256 * 1600;
    float* xg1 = wsf; wsf += 2 * 256 * 1600;
    float* WT0 = wsf; wsf += 2 * 400 * 1600;
    float* WT1 = wsf; wsf += 2 * 400 * 1600;
    float* Ap  = wsf; wsf += 256 * 1600;
    float* Bp  = wsf; wsf += 256 * 1600;
    float* lab = wsf; wsf += 256 * 40;
    __bf16* wsb   = (__bf16*)wsf;
    __bf16* xb    = wsb; wsb += 256 * 400;
    __bf16* Wih0b = wsb; wsb += 2 * 1600 * 400;
    __bf16* Wih1b = wsb; wsb += 2 * 1600 * 800;
    __bf16* Wa1b  = wsb; wsb += 1600 * 1600;
    __bf16* Wl1b  = wsb; wsb += 1600 * 1600;
    __bf16* Wl2b  = wsb; wsb += 40 * 1600;
    __bf16* hs0b  = wsb; wsb += 256 * 800;
    __bf16* hs1b  = wsb; wsb += 256 * 800;
    __bf16* lfb   = wsb; wsb += 256 * 1600;
    __bf16* lhidb = wsb; wsb += 256 * 1600;

    // one-time weight prep
    k_f2b<<<(2 * 1600 * 400 + 255) / 256, 256, 0, stream>>>(Wih0, Wih0b, 2 * 1600 * 400);
    k_f2b<<<(2 * 1600 * 800 + 255) / 256, 256, 0, stream>>>(Wih1, Wih1b, 2 * 1600 * 800);
    k_f2b<<<(1600 * 1600 + 255) / 256, 256, 0, stream>>>(Wa1, Wa1b, 1600 * 1600);
    k_f2b<<<(1600 * 1600 + 255) / 256, 256, 0, stream>>>(Wl1, Wl1b, 1600 * 1600);
    k_f2b<<<(40 * 1600 + 255) / 256, 256, 0, stream>>>(Wl2, Wl2b, 40 * 1600);
    k_transpose<<<dim3(2500, 2), 256, 0, stream>>>(Whh0, WT0, 1600, 400);
    k_transpose<<<dim3(2500, 2), 256, 0, stream>>>(Whh1, WT1, 1600, 400);
    k_embed<<<256, 512, 0, stream>>>(words, tags, wemb, temb, xb);

    // layer 0: xg = x @ Wih0[d]^T + (bih0[d]+bhh0[d]), then recurrence
    for (int d = 0; d < 2; ++d)
        k_gemm<<<dim3(16, 4), 256, 0, stream>>>(xb, 400,
            Wih0b + (long)d * 1600 * 400, 400,
            bih0 + d * 1600, bhh0 + d * 1600,
            xg0 + (long)d * 256 * 1600, nullptr, 1600, 1600, 400, 0);
    k_lstm<<<2, 1024, 0, stream>>>(xg0, WT0, h0, c0, 0, hs0b);

    // layer 1
    for (int d = 0; d < 2; ++d)
        k_gemm<<<dim3(16, 4), 256, 0, stream>>>(hs0b, 800,
            Wih1b + (long)d * 1600 * 800, 800,
            bih1 + d * 1600, bhh1 + d * 1600,
            xg1 + (long)d * 256 * 1600, nullptr, 1600, 1600, 800, 0);
    k_lstm<<<2, 1024, 0, stream>>>(xg1, WT1, h0, c0, 1, hs1b);

    // biaffine factorization of the arc MLP first layer
    k_gemm<<<dim3(16, 4), 256, 0, stream>>>(hs1b, 800, Wa1b,       1600, ba1,     nullptr, Ap, nullptr, 1600, 1600, 800, 0);
    k_gemm<<<dim3(16, 4), 256, 0, stream>>>(hs1b, 800, Wa1b + 800, 1600, nullptr, nullptr, Bp, nullptr, 1600, 1600, 800, 0);
    k_arc<<<256, 256, 0, stream>>>(Ap, Bp, Wa2, ba2, outF);
    k_border<<<2, 256, 0, stream>>>(outF);

    // label head
    k_lf<<<256, 256, 0, stream>>>(hs1b, arcs, lfb);
    k_gemm<<<dim3(16, 4), 256, 0, stream>>>(lfb,   1600, Wl1b, 1600, bl1, nullptr, nullptr, lhidb, 1600, 1600, 1600, 1);
    k_gemm<<<dim3(16, 1), 256, 0, stream>>>(lhidb, 1600, Wl2b, 1600, bl2, nullptr, lab,     nullptr,  40,   40, 1600, 0);
    k_labfin<<<40, 256, 0, stream>>>(lab, arcs, outF + 257 * 257);
}
